// CostRegNetRED_51599737094661
// MI455X (gfx1250) — compile-verified, matrix operand order checked by reading, not run on hardware
//
#include <hip/hip_runtime.h>
#include <hip/hip_bf16.h>

typedef __attribute__((ext_vector_type(16))) _Float16 v16h;
typedef __attribute__((ext_vector_type(8)))  _Float16 v8h;
typedef __attribute__((ext_vector_type(8)))  float    v8f;

__device__ __forceinline__ float act_apply(float x, int act) {
    if (act == 1) return x > 0.f ? x : 0.f;                      // relu
    if (act == 2) return 1.f / (1.f + __expf(-x));               // sigmoid
    if (act == 3) { float e = __expf(2.f * x); return (e - 1.f) / (e + 1.f); } // tanh
    return x;
}

// ---------------------------------------------------------------------------
// Two-phase LDS-staged implicit-GEMM 3x3 conv (pad=1, stride 1|2) on
// v_wmma_f32_16x16x32_f16.  Input = concat(src1[C1], src2[C2]*(mul2?:1)).
// Block: 160 threads = 5 waves; one 80-pixel output row segment x 16-ch N tile.
//   phase 1: patch[C][3][Wp] f16 from global (zero halo), Wp = 80*stride+2
//            wldsT[16][Kpad] f16  (N-transposed weight sub-tile)
//   phase 2: A_lds[80][Kpad] f16  im2col expansion (padding taps = 0),
//            vectorized ds_store_b128 writes.
//   compute: per K-chunk of 32: 2x ds_load_b128 (A) + 2x ds_load_b128 (B)
//            + v_wmma.  No conditionals / divisions in the hot loop.
// K is tap-major: k = tap*C + ci, C in {8,16,32}; taps >= 9 are zeros.
// A lane layout (16-bit A 16x32): lane L -> M=L%16, kb=(L<16?0:8),
//   half j -> k_local = kb + (j<8 ? j : j+8).  B mirrored with N=L%16.
// C/D: vgpr r -> M=(L<16 ? r : r+8), N=L%16.
// ---------------------------------------------------------------------------
__global__ void conv3x3_wmma(
    const float* __restrict__ src1, int C1, int cs1,
    const float* __restrict__ src2, int C2, int cs2,
    const float* __restrict__ mul2,
    const _Float16* __restrict__ wf16, int Kpad, int Ncols, int lg2C,
    const float* __restrict__ bias, int Cout,
    float* __restrict__ out,
    int H, int W, int Ho, int Wo, int stride, int act)
{
    __shared__ _Float16 patch[7872];                                   // C*3*Wp max
    __shared__ __attribute__((aligned(16))) _Float16 A_lds[80 * 288];  // 80 x Kpad max
    __shared__ __attribute__((aligned(16))) _Float16 wldsT[16 * 288];  // 16 x Kpad max

    const int tid  = threadIdx.x;
    const int lane = tid & 31;
    const int wv   = tid >> 5;                 // 0..4
    const int wob  = Wo / 80;                  // blocks per output row (exact)
    const int oy   = blockIdx.x / wob;
    const int ox0  = (blockIdx.x - oy * wob) * 80;
    const int n0   = blockIdx.y << 4;
    const int C    = C1 + C2;
    const int Wp   = 80 * stride + 2;
    const int ixb  = ox0 * stride - 1;

    // ---- phase 1: stage raw patch (f32 -> f16, zero halo, fused r*h mul) ----
    for (int ci = 0; ci < C; ++ci) {
        const float* sp;
        const float* mp = nullptr;
        if (ci < C1) {
            sp = src1 + (size_t)ci * cs1;
        } else {
            sp = src2 + (size_t)(ci - C1) * cs2;
            if (mul2) mp = mul2 + (size_t)(ci - C1) * cs2;
        }
        for (int r = 0; r < 3; ++r) {
            const int iy = oy * stride + r - 1;
            const bool rowok = (iy >= 0) && (iy < H);
            if (rowok && tid == 0)
                __builtin_prefetch(sp + (size_t)iy * W + (ixb < 0 ? 0 : ixb), 0, 0);
            _Float16* dst = patch + (ci * 3 + r) * Wp;
            for (int c = tid; c < Wp; c += 160) {
                const int ix = ixb + c;
                float v = 0.f;
                if (rowok && ix >= 0 && ix < W) {
                    v = sp[(size_t)iy * W + ix];
                    if (mp) v *= mp[(size_t)iy * W + ix];
                }
                dst[c] = (_Float16)v;
            }
        }
    }
    // ---- stage weight sub-tile, transposed to [n][Kpad] ----
    for (int n = 0; n < 16; ++n)
        for (int k = tid; k < Kpad; k += 160)
            wldsT[n * Kpad + k] = wf16[k * Ncols + n0 + n];

    __syncthreads();

    // ---- phase 2: im2col expansion into A_lds[pix][Kpad] ----
    {
        const int pix   = tid >> 1;            // 0..79
        const int tp0   = tid & 1;
        const int ntaps = Kpad >> lg2C;        // 9, 10 or 12
        _Float16* adst  = A_lds + pix * Kpad;
        const int colb  = pix * stride;
        for (int tap = tp0; tap < ntaps; tap += 2) {
            _Float16* drow = adst + (tap << lg2C);
            if (tap < 9) {
                const int ky = tap / 3;
                const int kx = tap - 3 * ky;
                const _Float16* srow = patch + ky * Wp + colb + kx;
                for (int c0 = 0; c0 < C; c0 += 8) {
                    v8h t;
#pragma unroll
                    for (int q = 0; q < 8; ++q)
                        t[q] = srow[(c0 + q) * 3 * Wp];
                    *(v8h*)(drow + c0) = t;
                }
            } else {
                const v8h z = {};
                for (int c0 = 0; c0 < C; c0 += 8)
                    *(v8h*)(drow + c0) = z;
            }
        }
    }

    __syncthreads();

    // ---- WMMA main loop: wide DS loads, no address math ----
    const int mm = lane & 15;
    const int kb = (lane < 16) ? 0 : 8;
    const _Float16* arow = A_lds + (wv * 16 + mm) * Kpad + kb;
    const _Float16* brow = wldsT + mm * Kpad + kb;

    v8f acc = {};
    for (int kc = 0; kc < Kpad; kc += 32) {
        const v8h a0 = *(const v8h*)(arow + kc);        // k = kc+kb+0..7
        const v8h a1 = *(const v8h*)(arow + kc + 16);   // k = kc+kb+16..23
        const v8h b0 = *(const v8h*)(brow + kc);
        const v8h b1 = *(const v8h*)(brow + kc + 16);
        const v16h a = __builtin_shufflevector(a0, a1, 0,1,2,3,4,5,6,7,8,9,10,11,12,13,14,15);
        const v16h b = __builtin_shufflevector(b0, b1, 0,1,2,3,4,5,6,7,8,9,10,11,12,13,14,15);
        acc = __builtin_amdgcn_wmma_f32_16x16x32_f16(
            /*neg_a=*/false, a, /*neg_b=*/false, b,
            /*c_mod=*/(short)0, acc, /*reuse_a=*/false, /*reuse_b=*/false);
    }

    // ---- bias + activation + store ----
    const int oc = n0 + mm;
    if (oc < Cout) {
        const float bs = bias ? bias[oc] : 0.f;
        float* op = out + (size_t)oc * Ho * Wo + (size_t)oy * Wo + ox0 + wv * 16;
#pragma unroll
        for (int r = 0; r < 8; ++r) {
            const int m = (lane < 16) ? r : r + 8;
            op[m] = act_apply(acc[r] + bs, act);
        }
    }
}

// Pack f32 OIHW weights -> f16 [Kpad x Ncols], tap-major K: k = tap*C + ci.
__global__ void pack_w(const float* __restrict__ w, _Float16* __restrict__ dst,
                       int Cin, int Cout, int lg2C, int Kpad, int Ncols)
{
    const int idx = blockIdx.x * blockDim.x + threadIdx.x;
    if (idx >= Kpad * Ncols) return;
    const int k = idx / Ncols;
    const int n = idx - k * Ncols;
    const int tap = k >> lg2C;
    const int ci  = k & ((1 << lg2C) - 1);
    float v = 0.f;
    if (tap < 9 && ci < Cin && n < Cout)
        v = w[(n * Cin + ci) * 9 + tap];
    dst[idx] = (_Float16)v;
}

__global__ void zero_f32(float* __restrict__ p, int n)
{
    const int i = blockIdx.x * blockDim.x + threadIdx.x;
    if (i < n) p[i] = 0.f;
}

// GRU state update: h = u*h + (1-u)*c with u = g[C+c] channel (g has 2C chans).
__global__ void gru_update(const float* __restrict__ g, const float* __restrict__ cand,
                           float* __restrict__ h, int C, int HW)
{
    const int idx = blockIdx.x * blockDim.x + threadIdx.x;
    if (idx >= C * HW) return;
    const int c = idx / HW;
    const float u = g[(C + c) * HW + (idx - c * HW)];
    h[idx] = u * h[idx] + (1.f - u) * cand[idx];
}

// ConvTranspose2d k=3, s=2, p=1, output_padding=1 (gather form):
// out[o,oy,ox] = b[o] + sum x[i,iy,ix] * w[i,o,ky,kx]  where oy = 2*iy-1+ky.
__global__ void deconv_s2(const float* __restrict__ x, const float* __restrict__ w,
                          const float* __restrict__ bias, const float* __restrict__ add,
                          float* __restrict__ out, int Cin, int Cout,
                          int Hin, int Win, int do_relu)
{
    const int Hout = Hin * 2, Wout = Win * 2;
    const int idx = blockIdx.x * blockDim.x + threadIdx.x;
    if (idx >= Cout * Hout * Wout) return;
    const int o  = idx / (Hout * Wout);
    const int p  = idx - o * (Hout * Wout);
    const int oy = p / Wout;
    const int ox = p - oy * Wout;

    float acc = bias[o];
    for (int ky = 0; ky < 3; ++ky) {
        const int ty = oy + 1 - ky;
        if (ty < 0 || (ty & 1)) continue;
        const int iy = ty >> 1;
        if (iy >= Hin) continue;
        for (int kx = 0; kx < 3; ++kx) {
            const int tx = ox + 1 - kx;
            if (tx < 0 || (tx & 1)) continue;
            const int ix = tx >> 1;
            if (ix >= Win) continue;
            const float* xp = x + (size_t)iy * Win + ix;
            const float* wp = w + (o * 3 + ky) * 3 + kx;
            for (int i = 0; i < Cin; ++i)
                acc = fmaf(xp[(size_t)i * Hin * Win], wp[(size_t)i * Cout * 9], acc);
        }
    }
    if (add) acc += add[idx];
    if (do_relu) acc = acc > 0.f ? acc : 0.f;
    out[idx] = acc;
}

extern "C" void kernel_launch(void* const* d_in, const int* in_sizes, int n_in,
                              void* d_out, int out_size, void* d_ws, size_t ws_size,
                              hipStream_t stream)
{
    const float* vol     = (const float*)d_in[0];
    const float* conv1_w = (const float*)d_in[1];
    const float* conv1_b = (const float*)d_in[2];
    const float* gru1_wg = (const float*)d_in[3];
    const float* gru1_bg = (const float*)d_in[4];
    const float* gru1_wc = (const float*)d_in[5];
    const float* gru1_bc = (const float*)d_in[6];
    const float* conv2_w = (const float*)d_in[7];
    const float* conv2_b = (const float*)d_in[8];
    const float* gru2_wg = (const float*)d_in[9];
    const float* gru2_bg = (const float*)d_in[10];
    const float* gru2_wc = (const float*)d_in[11];
    const float* gru2_bc = (const float*)d_in[12];
    const float* up1_w   = (const float*)d_in[13];
    const float* up1_b   = (const float*)d_in[14];
    const float* up2_w   = (const float*)d_in[15];
    const float* up2_b   = (const float*)d_in[16];
    float* out = (float*)d_out;

    const int D = 48, H = 256, W = 320, H2 = 128, W2 = 160;
    const int HW = H * W, H2W2 = H2 * W2;

    // ---- workspace layout: f16 packed weights, then f32 activations ----
    _Float16* w16  = (_Float16*)d_ws;
    _Float16* wc1  = w16 + 0;       // conv1   : C=32, Kpad=288, N=16
    _Float16* wg1  = w16 + 4608;    // gru1_wg : C=16, Kpad=160, N=16
    _Float16* wcd1 = w16 + 7168;    // gru1_wc : C=16, Kpad=160, N=16
    _Float16* wcv2 = w16 + 9728;    // conv2   : C=8,  Kpad=96,  N=16
    _Float16* wg2  = w16 + 11264;   // gru2_wg : C=32, Kpad=288, N=32
    _Float16* wcd2 = w16 + 20480;   // gru2_wc : C=32, Kpad=288, N=16

    float* f = (float*)((char*)d_ws + 65536);
    size_t o = 0;
    float* s1  = f + o;  o += (size_t)8  * HW;
    float* s2  = f + o;  o += (size_t)16 * H2W2;
    float* c1  = f + o;  o += (size_t)8  * HW;
    float* g1  = f + o;  o += (size_t)16 * HW;
    float* cd1 = f + o;  o += (size_t)8  * HW;
    float* c2  = f + o;  o += (size_t)16 * H2W2;
    float* g2  = f + o;  o += (size_t)32 * H2W2;
    float* cd2 = f + o;  o += (size_t)16 * H2W2;
    float* ub  = f + o;  o += (size_t)8  * HW;

    // ---- pack weights once (tap-major K) ----
    auto pack = [&](const float* w, _Float16* dst, int Cin, int Cout, int lg2C,
                    int Kpad, int Ncols) {
        const int total = Kpad * Ncols;
        pack_w<<<(total + 255) / 256, 256, 0, stream>>>(w, dst, Cin, Cout, lg2C, Kpad, Ncols);
    };
    pack(conv1_w, wc1,  32, 8,  5, 288, 16);
    pack(gru1_wg, wg1,  16, 16, 4, 160, 16);
    pack(gru1_wc, wcd1, 16, 8,  4, 160, 16);
    pack(conv2_w, wcv2, 8,  16, 3, 96,  16);
    pack(gru2_wg, wg2,  32, 32, 5, 288, 32);
    pack(gru2_wc, wcd2, 32, 16, 5, 288, 16);

    // ---- zero initial GRU states (s1, s2 contiguous at f) ----
    {
        const int n = 8 * HW + 16 * H2W2;
        zero_f32<<<(n + 255) / 256, 256, 0, stream>>>(s1, n);
    }

    auto conv = [&](const float* p1, int C1, int cs1,
                    const float* p2, int C2, int cs2, const float* mulp,
                    const _Float16* wptr, int Kpad, int Ncols, int lg2C,
                    const float* bptr, int Cout, float* outp,
                    int Hi, int Wi, int Ho, int Wo, int st, int act) {
        dim3 grid(Ho * (Wo / 80), Ncols / 16, 1);
        conv3x3_wmma<<<grid, 160, 0, stream>>>(p1, C1, cs1, p2, C2, cs2, mulp,
                                               wptr, Kpad, Ncols, lg2C, bptr, Cout,
                                               outp, Hi, Wi, Ho, Wo, st, act);
    };

    for (int d = 0; d < D; ++d) {
        const float* xslice = vol + (size_t)d * HW;     // channel stride D*HW

        // c1 = relu(conv1(x))
        conv(xslice, 32, D * HW, nullptr, 0, 0, nullptr,
             wc1, 288, 16, 5, conv1_b, 8, c1, H, W, H, W, 1, 1);
        // g1 = sigmoid(conv(cat(c1, s1), wg1))
        conv(c1, 8, HW, s1, 8, HW, nullptr,
             wg1, 160, 16, 4, gru1_bg, 16, g1, H, W, H, W, 1, 2);
        // cd1 = tanh(conv(cat(c1, r1*s1), wc1)) ; r1 = g1[0:8]
        conv(c1, 8, HW, s1, 8, HW, g1,
             wcd1, 160, 16, 4, gru1_bc, 8, cd1, H, W, H, W, 1, 3);
        // s1 = u1*s1 + (1-u1)*cd1 ; u1 = g1[8:16]
        gru_update<<<(8 * HW + 255) / 256, 256, 0, stream>>>(g1, cd1, s1, 8, HW);

        // c2 = relu(conv2(s1, stride 2))
        conv(s1, 8, HW, nullptr, 0, 0, nullptr,
             wcv2, 96, 16, 3, conv2_b, 16, c2, H, W, H2, W2, 2, 1);
        // g2 = sigmoid(conv(cat(c2, s2), wg2))
        conv(c2, 16, H2W2, s2, 16, H2W2, nullptr,
             wg2, 288, 32, 5, gru2_bg, 32, g2, H2, W2, H2, W2, 1, 2);
        // cd2 = tanh(conv(cat(c2, r2*s2), wc2)) ; r2 = g2[0:16]
        conv(c2, 16, H2W2, s2, 16, H2W2, g2,
             wcd2, 288, 16, 5, gru2_bc, 16, cd2, H2, W2, H2, W2, 1, 3);
        // s2 = u2*s2 + (1-u2)*cd2 ; u2 = g2[16:32]
        gru_update<<<(16 * H2W2 + 255) / 256, 256, 0, stream>>>(g2, cd2, s2, 16, H2W2);

        // ub = relu(deconv(s2, up1_w) + s1)   : (8, H, W)
        deconv_s2<<<(8 * HW + 255) / 256, 256, 0, stream>>>(
            s2, up1_w, up1_b, s1, ub, 16, 8, H2, W2, 1);
        // out[d] = deconv(ub, up2_w)          : (1, 2H, 2W)
        deconv_s2<<<(4 * HW + 255) / 256, 256, 0, stream>>>(
            ub, up2_w, up2_b, nullptr, out + (size_t)d * 4 * HW, 8, 1, H, W, 0);
    }
}
